// MultiHeadSelfAttention_50233937494537
// MI455X (gfx1250) — matrix operand order repaired and checked by reading, not compiled, not run on hardware
//
#include <hip/hip_runtime.h>

// MI455X / gfx1250 multi-head causal self-attention (flash-attention style).
//
// Roofline: attention math (2*B*H*T^2*HD*2 ~ 69 GFLOP) dominates; mandatory
// HBM traffic is only ~72 MB (~3 us at 23.3 TB/s), so the kernel is matrix-
// compute bound -> all GEMMs run through v_wmma_f32_16x16x32_f16 (f16 in,
// f32 accumulate; logits are sums of 64 ~N(0,1) products scaled by 1/8, well
// inside f16 range). S (B*H*T^2 = 1 GB) is never materialized (online
// softmax). K/V tiles are staged into LDS via the Tensor Data Mover
// (tensor_load_to_lds + s_wait_tensorcnt) and shared by 4 waves / 128 query
// rows per workgroup. Weights are converted to f16 once per launch so the
// GEMM hot loops carry no v_cvt traffic.
//
// Workspace (f16): Q | K | V | attn-out | Wp16 | Wq16t  ~ 66.5 MB.

typedef __attribute__((ext_vector_type(16))) _Float16 v16h;
typedef __attribute__((ext_vector_type(4)))  _Float16 v4h;
typedef __attribute__((ext_vector_type(8)))  float    v8f;
typedef __attribute__((ext_vector_type(4)))  unsigned int u32x4;
typedef __attribute__((ext_vector_type(8)))  int      i32x8;
typedef __attribute__((ext_vector_type(4)))  int      i32x4;

constexpr int Bn = 4, Tn = 2048, En = 1024, Hn = 16, HDn = 64;
constexpr int QKVN = 3 * HDn;   // 192
constexpr int QROWS = 128;      // query rows per workgroup (4 waves x 32)

// Tensor Data Mover availability (ROCm 7.2: 5-arg builtin; therock: 6-arg).
#if defined(__has_builtin)
#  if __has_builtin(__builtin_amdgcn_tensor_load_to_lds) && \
      __has_builtin(__builtin_amdgcn_s_wait_tensorcnt)
#    define USE_TDM 1
#  endif
#endif
#ifndef USE_TDM
#  define USE_TDM 0
#endif

// A-matrix (16x32 f16) K index for element e of the v16h fragment, per ISA
// 7.12.2: lanes 0-15 -> VGPR0-3 hold K=0..7, VGPR4-7 hold K=16..23;
// lanes 16-31 add +8.
__device__ __forceinline__ int a_kidx(int e, int half) {
  int v = e >> 1;
  int base = (v < 4) ? (2 * v) : (16 + 2 * (v - 4));
  return base + (e & 1) + 8 * half;
}
// B-matrix (32x16 f16) K index: same K order as the A fragment;
// lanes 0-15 hold K=0..7,16..23, lanes 16-31 add +8.
__device__ __forceinline__ int b_kidx(int e, int half) {
  return a_kidx(e, half);
}

#if USE_TDM
// Issue a TDM load of a 32x64 f16 2D tile (row stride 64 elem) from a
// [Tn, 64] f16 tensor into LDS.  D# per CDNA5 ISA ch.8:
//  group0: count=1 | lds_addr | global_addr[56:0] | type=2
//  group1: data_size=1(2B), tensor_dim0=64, tensor_dim1=Tn, tile_dim0=64,
//          tile_dim1=32, tensor_dim0_stride=64, no padding/iterate/multicast.
__device__ __forceinline__ void tdm_load_tile_32x64(unsigned lds_addr,
                                                    unsigned long long gaddr) {
  u32x4 g0;
  g0[0] = 1u;                                              // count=1, user D#
  g0[1] = lds_addr;                                        // LDS byte address
  g0[2] = (unsigned)(gaddr & 0xffffffffull);               // global_addr lo
  g0[3] = (unsigned)((gaddr >> 32) & 0x1ffffffull)         // global_addr hi
          | (2u << 30);                                    // type = 2 (image)
  i32x8 g1;
  g1[0] = (int)(1u << 16);            // wg_mask=0, data_size=1 (2 bytes)
  g1[1] = (int)(64u << 16);           // tensor_dim0 = 64 (bits 79:48 lo)
  g1[2] = (int)((unsigned)Tn << 16);  // tensor_dim1 = 2048 (bits 111:80 lo)
  g1[3] = (int)(64u << 16);           // tile_dim0 = 64 (bits 127:112)
  g1[4] = 32;                         // tile_dim1 = 32, tile_dim2 = 0
  g1[5] = 64;                         // tensor_dim0_stride = 64 (bits 207:160)
  g1[6] = 0;                          // stride hi, tensor_dim1_stride lo
  g1[7] = 0;
  i32x4 z4 = {0, 0, 0, 0};            // groups 2/3 unused (2D tensor)
#if __has_include(<hip/amd_detail/amd_gfx1250_TDM.h>)
  i32x8 z8 = {0, 0, 0, 0, 0, 0, 0, 0};
  __builtin_amdgcn_tensor_load_to_lds(g0, g1, z4, z4, z8, 0);
#else
  __builtin_amdgcn_tensor_load_to_lds(g0, g1, z4, z4, 0);
#endif
}
#endif  // USE_TDM

// ---------------------------------------------------------------------------
// Kernel 0a: W_proj fp32 -> f16, same [E,E] layout (row j contiguous in K).
// ---------------------------------------------------------------------------
__global__ void __launch_bounds__(256)
convert_wproj_kernel(const float* __restrict__ Wp, _Float16* __restrict__ Wp16) {
  const int i = blockIdx.x * 256 + threadIdx.x;   // over E*E/4
  const float4 v = ((const float4*)Wp)[i];
  v4h o;
  o[0] = (_Float16)v.x; o[1] = (_Float16)v.y;
  o[2] = (_Float16)v.z; o[3] = (_Float16)v.w;
  ((v4h*)Wp16)[i] = o;
}

// ---------------------------------------------------------------------------
// Kernel 0b: W_qkv fp32 [H,HD,192] -> f16 transposed [H,192,HD] so the QKV
// B-fragments become K-contiguous (b128 loads, no strided gathers).
// ---------------------------------------------------------------------------
__global__ void __launch_bounds__(256)
convert_wqkv_kernel(const float* __restrict__ Wq, _Float16* __restrict__ Wq16t) {
  const int idx = blockIdx.x * 256 + threadIdx.x;   // over H*HD*QKVN
  const int h  = idx / (HDn * QKVN);
  const int r  = idx % (HDn * QKVN);
  const int k  = r / QKVN;
  const int nc = r % QKVN;
  Wq16t[((size_t)h * QKVN + nc) * HDn + k] = (_Float16)Wq[idx];
}

// ---------------------------------------------------------------------------
// Kernel 1: per-head QKV projection.
// grid = (B*T/16, H), block = 32.  Each wave: a 16x192 slab of the per-head
// QKV output (12 N-tiles share one A fragment per K-chunk).  K = 64 -> 2
// chunks of 32.  Q pre-scaled by 1/(sqrt(HD)+1e-5); stored f16 [B,H,T,HD].
// ---------------------------------------------------------------------------
__global__ void __launch_bounds__(32)
qkv_proj_kernel(const float* __restrict__ x,
                const _Float16* __restrict__ Wq16t,   // [H,192,HD] f16
                const float* __restrict__ bqkv,
                _Float16* __restrict__ Qo,
                _Float16* __restrict__ Ko,
                _Float16* __restrict__ Vo) {
  const int lane = threadIdx.x & 31;
  const int half = lane >> 4;
  const int n    = lane & 15;
  const int row0 = blockIdx.x * 16;     // over B*T
  const int h    = blockIdx.y;

  v8f acc[12] = {};
#pragma unroll
  for (int kc = 0; kc < 2; ++kc) {      // K = 64 in chunks of 32
    v16h a;
    const float* xrow = x + (size_t)(row0 + n) * En + h * HDn + kc * 32;
#pragma unroll
    for (int e = 0; e < 16; ++e) a[e] = (_Float16)xrow[a_kidx(e, half)];
#pragma unroll
    for (int nt = 0; nt < 12; ++nt) {
      v16h b;
      const _Float16* wcol =
          Wq16t + ((size_t)h * QKVN + nt * 16 + n) * HDn + kc * 32;
#pragma unroll
      for (int e = 0; e < 16; ++e) b[e] = wcol[b_kidx(e, half)];
      acc[nt] = __builtin_amdgcn_wmma_f32_16x16x32_f16(false, a, false, b,
                                                       (short)0, acc[nt],
                                                       false, false);
    }
  }

  const float scale = 1.0f / (8.0f + 1e-5f);   // 1/(sqrt(64)+1e-5)
#pragma unroll
  for (int nt = 0; nt < 12; ++nt) {
    const int which = nt >> 2;                  // 0=Q 1=K 2=V
    const int hd    = (nt & 3) * 16 + n;
    const float bias = bqkv[h * QKVN + nt * 16 + n];
#pragma unroll
    for (int r = 0; r < 8; ++r) {
      const int grow = row0 + r + 8 * half;     // index over B*T
      const int bb = grow / Tn, tt = grow % Tn;
      const size_t idx = (((size_t)bb * Hn + h) * Tn + tt) * HDn + hd;
      const float val = acc[nt][r] + bias;
      if (which == 0)      Qo[idx] = (_Float16)(val * scale);
      else if (which == 1) Ko[idx] = (_Float16)val;
      else                 Vo[idx] = (_Float16)val;
    }
  }
}

// ---------------------------------------------------------------------------
// Kernel 2: causal flash attention.
// grid = (T/128, B*H), block = 128 (4 waves).  Wave w owns query rows
// [q0+32w, q0+32w+31] as two 16-row subtiles.  Key loop in 32-wide tiles:
// K/V staged into LDS by the Tensor Data Mover (wave 0 issues, TENSORcnt
// waited, then workgroup barrier), S via 4 WMMAs per subtile, online softmax
// with shfl-xor reductions in 16-lane groups, P transposed C->A layout
// through per-wave LDS buffers, O += P*V (4 WMMAs per subtile).  Control
// flow is wave-uniform everywhere (EXEC all ones at WMMA; barriers
// non-divergent); fully-masked rows contribute exact zeros.
// ---------------------------------------------------------------------------
__global__ void __launch_bounds__(128)
flash_attn_kernel(const _Float16* __restrict__ Qm,
                  const _Float16* __restrict__ Km,
                  const _Float16* __restrict__ Vm,
                  _Float16* __restrict__ Om) {
  const int tid  = threadIdx.x;
  const int lane = tid & 31;
  const int w    = tid >> 5;            // wave 0..3
  const int half = lane >> 4;
  const int n    = lane & 15;
  const int bh   = blockIdx.y;
  const size_t base = (size_t)bh * Tn * HDn;
  const int q0  = blockIdx.x * QROWS;
  const int wq0 = q0 + w * 32;          // this wave's first query row

  __shared__ _Float16 Kt[32 * HDn];     // 4 KB
  __shared__ _Float16 Vt[32 * HDn];     // 4 KB
  __shared__ _Float16 Pt[4][2][16 * 32];// 8 KB: per wave, per row-subtile

  // Q fragments: 2 subtiles x (K=64 -> 2 fragments), held for whole loop.
  v16h aq[2][2];
#pragma unroll
  for (int u = 0; u < 2; ++u) {
    const _Float16* qrow = Qm + base + (size_t)(wq0 + u * 16 + n) * HDn;
#pragma unroll
    for (int e = 0; e < 16; ++e) {
      const int kk = a_kidx(e, half);
      aq[u][0][e] = qrow[kk];
      aq[u][1][e] = qrow[kk + 32];
    }
  }

  v8f acc[2][4] = {};                   // O accumulators: 2 subtiles x 64 cols
  float rmax[2][8], rsum[2][8];
#pragma unroll
  for (int u = 0; u < 2; ++u)
#pragma unroll
    for (int r = 0; r < 8; ++r) { rmax[u][r] = -3.0e38f; rsum[u][r] = 0.0f; }

#if USE_TDM
  const unsigned ldsK =
      (unsigned)(size_t)(__attribute__((address_space(3))) void*)Kt;
  const unsigned ldsV =
      (unsigned)(size_t)(__attribute__((address_space(3))) void*)Vt;
#endif

  const int ktiles = (q0 + QROWS) / 32; // causal bound of the whole workgroup
  for (int j = 0; j < ktiles; ++j) {
    const int s0 = j * 32;

    // --- Stage K/V 32x64 f16 tiles into LDS. ---
#if USE_TDM
    if (w == 0) {                       // wave-uniform; TDM ignores EXEC
      tdm_load_tile_32x64(
          ldsK, (unsigned long long)(size_t)(Km + base + (size_t)s0 * HDn));
      tdm_load_tile_32x64(
          ldsV, (unsigned long long)(size_t)(Vm + base + (size_t)s0 * HDn));
      __builtin_amdgcn_s_wait_tensorcnt(0);
    }
#else
    {
      const int4* gk = (const int4*)(Km + base + (size_t)s0 * HDn);
      const int4* gv = (const int4*)(Vm + base + (size_t)s0 * HDn);
      int4* lk = (int4*)Kt;
      int4* lv = (int4*)Vt;
      lk[tid]       = gk[tid];
      lk[tid + 128] = gk[tid + 128];
      lv[tid]       = gv[tid];
      lv[tid + 128] = gv[tid + 128];
    }
#endif
    if (j + 1 < ktiles) {               // prefetch next tile (global_prefetch_b8)
      __builtin_prefetch(Km + base + (size_t)(s0 + 32) * HDn + (tid & 63) * 32, 0, 0);
      __builtin_prefetch(Vm + base + (size_t)(s0 + 32) * HDn + (tid & 63) * 32, 0, 0);
    }
    __syncthreads();

    // --- Per row-subtile: S = Q*K^T, mask, online softmax, P to LDS. ---
#pragma unroll
    for (int u = 0; u < 2; ++u) {
      v8f s[2];
#pragma unroll
      for (int c = 0; c < 2; ++c) {
        v16h bk0, bk1;
        const _Float16* krow = Kt + (c * 16 + n) * HDn;
#pragma unroll
        for (int e = 0; e < 16; ++e) {
          const int kk = a_kidx(e, half);
          bk0[e] = krow[kk];
          bk1[e] = krow[kk + 32];
        }
        v8f z = {};
        z    = __builtin_amdgcn_wmma_f32_16x16x32_f16(false, aq[u][0], false,
                                                      bk0, (short)0, z,
                                                      false, false);
        s[c] = __builtin_amdgcn_wmma_f32_16x16x32_f16(false, aq[u][1], false,
                                                      bk1, (short)0, z,
                                                      false, false);
      }

      float tmax[8];
#pragma unroll
      for (int r = 0; r < 8; ++r) {
        const int tq = wq0 + u * 16 + r + 8 * half;
        const float v0 = (s0 + n      <= tq) ? s[0][r] : -3.0e38f;
        const float v1 = (s0 + 16 + n <= tq) ? s[1][r] : -3.0e38f;
        s[0][r] = v0; s[1][r] = v1;
        tmax[r] = fmaxf(v0, v1);
      }
#pragma unroll
      for (int m = 1; m < 16; m <<= 1)
#pragma unroll
        for (int r = 0; r < 8; ++r)
          tmax[r] = fmaxf(tmax[r], __shfl_xor(tmax[r], m, 32));

      float psum[8];
#pragma unroll
      for (int r = 0; r < 8; ++r) {
        const float nm = fmaxf(rmax[u][r], tmax[r]);
        const float corr = __expf(rmax[u][r] - nm);
        rmax[u][r] = nm;
        const float p0 = __expf(s[0][r] - nm);
        const float p1 = __expf(s[1][r] - nm);
        s[0][r] = p0; s[1][r] = p1;
        psum[r] = p0 + p1;
        rsum[u][r] *= corr;
#pragma unroll
        for (int t = 0; t < 4; ++t) acc[u][t][r] *= corr;
      }
#pragma unroll
      for (int m = 1; m < 16; m <<= 1)
#pragma unroll
        for (int r = 0; r < 8; ++r)
          psum[r] += __shfl_xor(psum[r], m, 32);
#pragma unroll
      for (int r = 0; r < 8; ++r) rsum[u][r] += psum[r];

#pragma unroll
      for (int r = 0; r < 8; ++r) {
        const int m = r + 8 * half;
        Pt[w][u][m * 32 + n]      = (_Float16)s[0][r];
        Pt[w][u][m * 32 + 16 + n] = (_Float16)s[1][r];
      }
    }
    __syncthreads();

    // --- O += P * V: V fragments read once, shared by both subtiles. ---
    v16h bv[4];
#pragma unroll
    for (int t = 0; t < 4; ++t)
#pragma unroll
      for (int e = 0; e < 16; ++e)
        bv[t][e] = Vt[b_kidx(e, half) * HDn + t * 16 + n];
#pragma unroll
    for (int u = 0; u < 2; ++u) {
      v16h ap;
#pragma unroll
      for (int e = 0; e < 16; ++e) ap[e] = Pt[w][u][n * 32 + a_kidx(e, half)];
#pragma unroll
      for (int t = 0; t < 4; ++t)
        acc[u][t] = __builtin_amdgcn_wmma_f32_16x16x32_f16(false, ap, false,
                                                           bv[t], (short)0,
                                                           acc[u][t],
                                                           false, false);
    }
    __syncthreads();
  }

  // Normalize and store f16 into [B,T,E] (heads concatenated).
  const int bb = bh / Hn, hh = bh % Hn;
#pragma unroll
  for (int u = 0; u < 2; ++u)
#pragma unroll
    for (int r = 0; r < 8; ++r) {
      const int m = r + 8 * half;
      const float inv = 1.0f / rsum[u][r];
      _Float16* orow =
          Om + ((size_t)bb * Tn + (wq0 + u * 16 + m)) * En + hh * HDn;
#pragma unroll
      for (int t = 0; t < 4; ++t)
        orow[t * 16 + n] = (_Float16)(acc[u][t][r] * inv);
    }
}

// ---------------------------------------------------------------------------
// Kernel 3: output projection  out = A @ W_proj^T + b_proj  (fp32 out).
// grid = (B*T/16, E/64), block = 32.  Each wave: 16x64 output (4 N-tiles
// share one A fragment per K-chunk).  K loop: 1024/32 = 32 chunks.
// Wp16 row j contiguous in K -> B fragment loads merge to b128.
// ---------------------------------------------------------------------------
__global__ void __launch_bounds__(32)
out_proj_kernel(const _Float16* __restrict__ A,
                const _Float16* __restrict__ Wp16,
                const float* __restrict__ bp,
                float* __restrict__ out) {
  const int lane = threadIdx.x & 31;
  const int half = lane >> 4;
  const int n    = lane & 15;
  const int row0 = blockIdx.x * 16;
  const int col0 = blockIdx.y * 64;

  v8f acc[4] = {};
  for (int kc = 0; kc < En / 32; ++kc) {
    v16h a;
    const _Float16* arow = A + (size_t)(row0 + n) * En + kc * 32;
#pragma unroll
    for (int e = 0; e < 16; ++e) a[e] = arow[a_kidx(e, half)];
#pragma unroll
    for (int t = 0; t < 4; ++t) {
      v16h b;
      const _Float16* wrow = Wp16 + (size_t)(col0 + t * 16 + n) * En + kc * 32;
#pragma unroll
      for (int e = 0; e < 16; ++e) b[e] = wrow[b_kidx(e, half)];
      acc[t] = __builtin_amdgcn_wmma_f32_16x16x32_f16(false, a, false, b,
                                                      (short)0, acc[t],
                                                      false, false);
    }
  }

#pragma unroll
  for (int t = 0; t < 4; ++t) {
    const float bias = bp[col0 + t * 16 + n];
#pragma unroll
    for (int r = 0; r < 8; ++r)
      out[(size_t)(row0 + r + 8 * half) * En + col0 + t * 16 + n] =
          acc[t][r] + bias;
  }
}

// ---------------------------------------------------------------------------
extern "C" void kernel_launch(void* const* d_in, const int* in_sizes, int n_in,
                              void* d_out, int out_size, void* d_ws, size_t ws_size,
                              hipStream_t stream) {
  const float* x    = (const float*)d_in[0];
  const float* Wqkv = (const float*)d_in[1];
  const float* bqkv = (const float*)d_in[2];
  const float* Wp   = (const float*)d_in[3];
  const float* bp   = (const float*)d_in[4];
  float* out = (float*)d_out;

  // Workspace partition (f16): Q | K | V | attn-out | Wp16 | Wq16t.
  const size_t elems = (size_t)Bn * Hn * Tn * HDn;   // 8,388,608
  _Float16* Q    = (_Float16*)d_ws;
  _Float16* K    = Q + elems;
  _Float16* V    = K + elems;
  _Float16* O    = V + elems;
  _Float16* Wp16 = O + elems;                         // E*E elements
  _Float16* Wq16 = Wp16 + (size_t)En * En;            // H*192*HD elements

  convert_wproj_kernel<<<dim3(En * En / 4 / 256), 256, 0, stream>>>(Wp, Wp16);
  convert_wqkv_kernel<<<dim3(Hn * HDn * QKVN / 256), 256, 0, stream>>>(Wqkv, Wq16);
  qkv_proj_kernel<<<dim3(Bn * Tn / 16, Hn), 32, 0, stream>>>(
      x, Wq16, bqkv, Q, K, V);
  flash_attn_kernel<<<dim3(Tn / QROWS, Bn * Hn), 128, 0, stream>>>(Q, K, V, O);
  out_proj_kernel<<<dim3(Bn * Tn / 16, En / 64), 32, 0, stream>>>(O, Wp16, bp, out);
}
